// SelfAttention_81879256530964
// MI455X (gfx1250) — matrix operand order repaired and checked by reading, not compiled, not run on hardware
//
#include <hip/hip_runtime.h>
#include <hip/hip_bf16.h>

// ---------------------------------------------------------------------------
// Self-attention forward, MI455X (gfx1250, wave32, WMMA bf16).
// B=8, L=1024, D=1024, H=8, DK=128.
// GEMMs on v_wmma_f32_16x16x32_bf16; global->LDS staging via async DMA
// (global_load_async_to_lds_b128, ASYNCcnt) with double buffering.
// ---------------------------------------------------------------------------

#define BB   8
#define LL   1024
#define DD   1024
#define HH   8
#define DKK  128
#define HDIM 1024   // H*DK
#define NEGV (-1e30f)

typedef __attribute__((ext_vector_type(16))) __bf16 v16bf;
typedef __attribute__((ext_vector_type(8)))  __bf16 v8bf;
typedef __attribute__((ext_vector_type(8)))  float  v8f;

// WMMA D = A(16x32 bf16) * B(32x16 bf16) + C(16x16 f32)
static __device__ __forceinline__ v8f wmma_bf16(v16bf a, v16bf b, v8f c) {
  return __builtin_amdgcn_wmma_f32_16x16x32_bf16(
      /*neg_a=*/false, a, /*neg_b=*/false, b,
      /*c_mod=*/(short)0, c, /*reuse_a=*/false, /*reuse_b=*/false);
}

// Async DMA: 16B global -> LDS, tracked by ASYNCcnt (no VGPR data path).
// LDS address = low 32 bits of the generic pointer (ISA 10.2: flat LDS
// aperture uses addr[31:0] as the wave-relative LDS address).
static __device__ __forceinline__ void async_load16(void* lds_dst, const void* gsrc) {
  unsigned lds_off = (unsigned)(unsigned long long)lds_dst;
  asm volatile("global_load_async_to_lds_b128 %0, %1, off"
               :: "v"(lds_off), "v"(gsrc) : "memory");
}

// A-matrix fragment (16x32, row-major [m][k] tile, stride in elements).
// lanes 0-15: K0..7 in v0-3, K16..23 in v4-7 ; lanes 16-31: K8..15, K24..31.
static __device__ __forceinline__ v16bf load_fragA(const __bf16* base, int stride, int lane) {
  const int r    = lane & 15;
  const int koff = (lane & 16) ? 8 : 0;
  const __bf16* p = base + r * stride + koff;
  v8bf lo = *(const v8bf*)(p);
  v8bf hi = *(const v8bf*)(p + 16);
  return __builtin_shufflevector(lo, hi, 0,1,2,3,4,5,6,7,8,9,10,11,12,13,14,15);
}

// B-matrix fragment (32x16, staged as [n][k] tile, k contiguous).
// lanes 0-15: K0..7 in v0-3, K16..23 in v4-7 ; lanes 16-31: K8..15, K24..31.
static __device__ __forceinline__ v16bf load_fragB(const __bf16* base, int stride, int lane) {
  const int r    = lane & 15;
  const int koff = (lane & 16) ? 8 : 0;
  const __bf16* p = base + r * stride + koff;
  v8bf lo = *(const v8bf*)(p);
  v8bf hi = *(const v8bf*)(p + 16);
  return __builtin_shufflevector(lo, hi, 0,1,2,3,4,5,6,7,8,9,10,11,12,13,14,15);
}

// ---------------------------------------------------------------------------
// 1) x (B,D,L) f32  ->  xt (B,L,D) bf16   (transpose + downconvert)
// ---------------------------------------------------------------------------
__global__ void k_transpose_x(const float* __restrict__ x, __bf16* __restrict__ xt) {
  size_t idx = (size_t)blockIdx.x * 256 + threadIdx.x;     // over B*D*L
  int l = (int)(idx % LL);
  int d = (int)((idx / LL) % DD);
  int b = (int)(idx / ((size_t)LL * DD));
  xt[((size_t)b * LL + l) * DD + d] = (__bf16)x[idx];
}

// ---------------------------------------------------------------------------
// 2) weights (G,R,C) f32 -> (G,C,R) bf16  (transpose last two dims)
// ---------------------------------------------------------------------------
__global__ void k_convert_wt(const float* __restrict__ w, __bf16* __restrict__ wt,
                             int R, int C) {
  size_t idx = (size_t)blockIdx.x * 256 + threadIdx.x;     // over G*R*C
  int c = (int)(idx % C);
  size_t t = idx / C;
  int r = (int)(t % R);
  int g = (int)(t / R);
  wt[((size_t)g * C + c) * R + r] = (__bf16)w[idx];
}

// ---------------------------------------------------------------------------
// 3) QKV projection: q/k = xt @ W  -> [b,h,l,dk] bf16 ; v stored transposed
//    vT[b,h,dk,l] bf16. Block: 256 thr = 8 waves, tile 64(M) x 128(N), K=1024.
//    Double-buffered async staging: 7 async 16B copies per thread per step.
// ---------------------------------------------------------------------------
__global__ __launch_bounds__(256) void k_qkv(
    const __bf16* __restrict__ xt,
    const __bf16* __restrict__ wqt, const __bf16* __restrict__ wkt,
    const __bf16* __restrict__ wvt,
    __bf16* __restrict__ qg, __bf16* __restrict__ kg, __bf16* __restrict__ vtg) {
  __shared__ __bf16 As[2][64 * 40];          // [m][k], padded stride 40
  __shared__ __bf16 Bs[2][3][128 * 40];      // [n][k] for Wq/Wk/Wv

  const int tid  = threadIdx.x;
  const int lane = tid & 31;
  const int wave = tid >> 5;
  const int bh   = blockIdx.y;
  const int b    = bh >> 3;
  const int h    = bh & (HH - 1);
  const int m0   = blockIdx.x * 64;
  const int mt   = wave & 3;              // m-tile of this wave
  const int nh   = wave >> 2;             // n half: n tiles nh*4..nh*4+3

  const __bf16* wts[3] = {wqt, wkt, wvt};
  v8f accq[4] = {}, acck[4] = {}, accv[4] = {};

  const int ai = tid >> 2, ajc = tid & 3;   // A staging chunk

  auto stage = [&](int buf, int ks) {
    const int d0 = ks * 32;
    async_load16(&As[buf][ai * 40 + ajc * 8],
                 &xt[((size_t)b * LL + m0 + ai) * DD + d0 + ajc * 8]);
#pragma unroll
    for (int m = 0; m < 3; ++m) {
#pragma unroll
      for (int e = 0; e < 2; ++e) {
        int c = tid + e * 256;
        int n = c >> 2, kc = c & 3;
        async_load16(&Bs[buf][m][n * 40 + kc * 8],
                     &wts[m][((size_t)h * DKK + n) * DD + d0 + kc * 8]);
      }
    }
  };

  stage(0, 0);
  for (int ks = 0; ks < DD / 32; ++ks) {
    const int cur = ks & 1;
    if (ks + 1 < DD / 32) {
      stage(cur ^ 1, ks + 1);
      asm volatile("s_wait_asynccnt 0x7" ::: "memory");  // current tile done
    } else {
      asm volatile("s_wait_asynccnt 0x0" ::: "memory");
    }
    __syncthreads();

    v16bf a = load_fragA(&As[cur][mt * 16 * 40], 40, lane);
#pragma unroll
    for (int t = 0; t < 4; ++t) {
      v16bf bq = load_fragB(&Bs[cur][0][(nh * 64 + t * 16) * 40], 40, lane);
      accq[t] = wmma_bf16(a, bq, accq[t]);
      v16bf bk = load_fragB(&Bs[cur][1][(nh * 64 + t * 16) * 40], 40, lane);
      acck[t] = wmma_bf16(a, bk, acck[t]);
      v16bf bv = load_fragB(&Bs[cur][2][(nh * 64 + t * 16) * 40], 40, lane);
      accv[t] = wmma_bf16(a, bv, accv[t]);
    }
    __syncthreads();
  }

  // C layout: VGPR r -> M = r (lanes 0-15) / 8+r (lanes 16-31), N = lane&15
  const int col   = lane & 15;
  const int rbase = (lane & 16) ? 8 : 0;
#pragma unroll
  for (int t = 0; t < 4; ++t) {
    const int dk = nh * 64 + t * 16 + col;
#pragma unroll
    for (int r = 0; r < 8; ++r) {
      const int l = m0 + mt * 16 + rbase + r;
      qg[((size_t)bh * LL + l) * DKK + dk] = (__bf16)accq[t][r];
      kg[((size_t)bh * LL + l) * DKK + dk] = (__bf16)acck[t][r];
    }
    // v transposed: 8 consecutive l per lane -> one 16B store
    v8bf vv;
#pragma unroll
    for (int r = 0; r < 8; ++r) vv[r] = (__bf16)accv[t][r];
    *(v8bf*)&vtg[((size_t)bh * DKK + dk) * LL + m0 + mt * 16 + rbase] = vv;
  }
}

// ---------------------------------------------------------------------------
// 4) Flash attention: per (b,h), 64 query rows per block (4 waves x 16 rows),
//    key blocks of 32, online softmax, heads -> [b,l,h*DK] bf16.
//    Double-buffered async staging: 8 async 16B copies per thread per block.
// ---------------------------------------------------------------------------
__global__ __launch_bounds__(128) void k_attn(
    const __bf16* __restrict__ qg, const __bf16* __restrict__ kg,
    const __bf16* __restrict__ vtg, const float* __restrict__ mask,
    __bf16* __restrict__ heads) {
  __shared__ __bf16 Ks[2][32 * 136];     // [key][dk], stride 136
  __shared__ __bf16 Vts[2][128 * 40];    // [dk][key], stride 40
  __shared__ __bf16 Ps[4][16 * 40];      // per-wave P tile [m][key]

  const int tid   = threadIdx.x;
  const int lane  = tid & 31;
  const int wave  = tid >> 5;
  const int bh    = blockIdx.y;
  const int b     = bh >> 3;
  const int h     = bh & 7;
  const int m0    = blockIdx.x * 64;
  const int q0    = m0 + wave * 16;
  const int col   = lane & 15;
  const int rbase = (lane & 16) ? 8 : 0;
  const float scale = 0.08838834764831845f;   // 1/sqrt(128)

  // q fragments: whole dk=128 held in registers (4 x A-frag), from global
  v16bf qf[4];
#pragma unroll
  for (int kd = 0; kd < 4; ++kd)
    qf[kd] = load_fragA(qg + ((size_t)bh * LL + q0) * DKK + kd * 32, DKK, lane);

  v8f acc[8] = {};
  float m_i[8], l_i[8];
#pragma unroll
  for (int r = 0; r < 8; ++r) { m_i[r] = -3.0e38f; l_i[r] = 0.f; }

  auto stage = [&](int buf, int kb) {
    const int k0 = kb * 32;
#pragma unroll
    for (int e = 0; e < 4; ++e) {     // K rows [key][dk]
      int c = tid + e * 128;
      int key = c >> 4, kc = c & 15;
      async_load16(&Ks[buf][key * 136 + kc * 8],
                   &kg[((size_t)bh * LL + k0 + key) * DKK + kc * 8]);
    }
#pragma unroll
    for (int e = 0; e < 4; ++e) {     // V^T tile [dk][key]
      int c = tid + e * 128;
      int dk = c >> 2, kc = c & 3;
      async_load16(&Vts[buf][dk * 40 + kc * 8],
                   &vtg[((size_t)bh * DKK + dk) * LL + k0 + kc * 8]);
    }
  };

  stage(0, 0);
  for (int kb = 0; kb < LL / 32; ++kb) {
    const int cur = kb & 1;
    if (kb + 1 < LL / 32) {
      stage(cur ^ 1, kb + 1);
      asm volatile("s_wait_asynccnt 0x8" ::: "memory");
    } else {
      asm volatile("s_wait_asynccnt 0x0" ::: "memory");
    }
    __syncthreads();
    const int k0 = kb * 32;

    // scores: two 16-key tiles, contract over dk (4 WMMA each)
    v8f s[2] = {};
#pragma unroll
    for (int nt = 0; nt < 2; ++nt) {
#pragma unroll
      for (int kd = 0; kd < 4; ++kd) {
        v16bf kf = load_fragB(&Ks[cur][nt * 16 * 136 + kd * 32], 136, lane);
        s[nt] = wmma_bf16(qf[kd], kf, s[nt]);
      }
    }

    // scale + key mask + online softmax (row reductions across 16-lane halves)
    const float mk0 = mask[(size_t)b * LL + k0 + col];
    const float mk1 = mask[(size_t)b * LL + k0 + 16 + col];
    float alpha[8];
#pragma unroll
    for (int r = 0; r < 8; ++r) {
      float s0 = s[0][r] * scale * mk0 + (1.f - mk0) * NEGV;
      float s1 = s[1][r] * scale * mk1 + (1.f - mk1) * NEGV;
      float mx = fmaxf(s0, s1);
      mx = fmaxf(mx, __shfl_xor(mx, 1, 32));
      mx = fmaxf(mx, __shfl_xor(mx, 2, 32));
      mx = fmaxf(mx, __shfl_xor(mx, 4, 32));
      mx = fmaxf(mx, __shfl_xor(mx, 8, 32));
      const float mnew = fmaxf(m_i[r], mx);
      const float a0 = __expf(m_i[r] - mnew);
      const float p0 = __expf(s0 - mnew);
      const float p1 = __expf(s1 - mnew);
      float rs = p0 + p1;
      rs += __shfl_xor(rs, 1, 32);
      rs += __shfl_xor(rs, 2, 32);
      rs += __shfl_xor(rs, 4, 32);
      rs += __shfl_xor(rs, 8, 32);
      l_i[r] = l_i[r] * a0 + rs;
      m_i[r] = mnew;
      alpha[r] = a0;
      s[0][r] = p0;
      s[1][r] = p1;
    }
#pragma unroll
    for (int nt2 = 0; nt2 < 8; ++nt2)
#pragma unroll
      for (int r = 0; r < 8; ++r) acc[nt2][r] *= alpha[r];

    // P: C-layout -> LDS [m][key] -> A-fragment (same-wave DS, kept in order)
    __bf16* pp = &Ps[wave][0];
#pragma unroll
    for (int r = 0; r < 8; ++r) {
      const int row = rbase + r;
      pp[row * 40 + col]      = (__bf16)s[0][r];
      pp[row * 40 + 16 + col] = (__bf16)s[1][r];
    }
    asm volatile("s_wait_dscnt 0x0" ::: "memory");
    v16bf pf = load_fragA(pp, 40, lane);

    // acc += P (16x32) @ V (32 keys x 128 dk): 8 n-tiles
#pragma unroll
    for (int nt2 = 0; nt2 < 8; ++nt2) {
      v16bf vf = load_fragB(&Vts[cur][nt2 * 16 * 40], 40, lane);
      acc[nt2] = wmma_bf16(pf, vf, acc[nt2]);
    }
    __syncthreads();
  }

  // finalize: divide by l_i, apply query mask, store heads bf16
#pragma unroll
  for (int r = 0; r < 8; ++r) {
    const int row = rbase + r;
    const float qm = mask[(size_t)b * LL + q0 + row];
    const float inv = qm / l_i[r];
#pragma unroll
    for (int nt2 = 0; nt2 < 8; ++nt2) {
      heads[((size_t)b * LL + q0 + row) * HDIM + h * DKK + nt2 * 16 + col] =
          (__bf16)(acc[nt2][r] * inv);
    }
  }
}

// ---------------------------------------------------------------------------
// 5) output projection: out[b,d,l] = sum_e heads[b,l,e] * Wo[e,d]
//    (Wo pre-transposed to [d][e]); f32 output stored transposed with 32B
//    vector stores. Double-buffered async staging (3 copies/thread/step).
// ---------------------------------------------------------------------------
__global__ __launch_bounds__(256) void k_out(
    const __bf16* __restrict__ heads, const __bf16* __restrict__ wot,
    float* __restrict__ out) {
  __shared__ __bf16 As[2][64 * 40];
  __shared__ __bf16 Bs[2][128 * 40];

  const int tid  = threadIdx.x;
  const int lane = tid & 31;
  const int wave = tid >> 5;
  const int b    = blockIdx.z;
  const int m0   = blockIdx.x * 64;     // l tile
  const int n0   = blockIdx.y * 128;    // d tile
  const int mt   = wave & 3;
  const int nh   = wave >> 2;

  const int ai = tid >> 2, ajc = tid & 3;

  auto stage = [&](int buf, int ks) {
    const int e0 = ks * 32;
    async_load16(&As[buf][ai * 40 + ajc * 8],
                 &heads[((size_t)b * LL + m0 + ai) * HDIM + e0 + ajc * 8]);
#pragma unroll
    for (int e = 0; e < 2; ++e) {
      int c = tid + e * 256;
      int n = c >> 2, kc = c & 3;
      async_load16(&Bs[buf][n * 40 + kc * 8],
                   &wot[((size_t)(n0 + n)) * DD + e0 + kc * 8]);
    }
  };

  v8f acc[4] = {};
  stage(0, 0);
  for (int ks = 0; ks < HDIM / 32; ++ks) {
    const int cur = ks & 1;
    if (ks + 1 < HDIM / 32) {
      stage(cur ^ 1, ks + 1);
      asm volatile("s_wait_asynccnt 0x3" ::: "memory");
    } else {
      asm volatile("s_wait_asynccnt 0x0" ::: "memory");
    }
    __syncthreads();

    v16bf a = load_fragA(&As[cur][mt * 16 * 40], 40, lane);
#pragma unroll
    for (int t = 0; t < 4; ++t) {
      v16bf bb = load_fragB(&Bs[cur][(nh * 64 + t * 16) * 40], 40, lane);
      acc[t] = wmma_bf16(a, bb, acc[t]);
    }
    __syncthreads();
  }

  const int col   = lane & 15;
  const int rbase = (lane & 16) ? 8 : 0;
#pragma unroll
  for (int t = 0; t < 4; ++t) {
    const int d = n0 + nh * 64 + t * 16 + col;
    float* p = out + ((size_t)b * DD + d) * LL + m0 + mt * 16 + rbase;
    *(v8f*)p = acc[t];   // 8 consecutive l -> 32B store
  }
}

// ---------------------------------------------------------------------------
extern "C" void kernel_launch(void* const* d_in, const int* in_sizes, int n_in,
                              void* d_out, int out_size, void* d_ws, size_t ws_size,
                              hipStream_t stream) {
  (void)in_sizes; (void)n_in; (void)out_size; (void)ws_size;
  const float* x    = (const float*)d_in[0];
  const float* mask = (const float*)d_in[1];
  const float* Wq   = (const float*)d_in[2];
  const float* Wk   = (const float*)d_in[3];
  const float* Wv   = (const float*)d_in[4];
  const float* Wo   = (const float*)d_in[5];
  float* out = (float*)d_out;

  // workspace carve-up (bf16, ~92 MB total)
  __bf16* p    = (__bf16*)d_ws;
  __bf16* xt   = p; p += (size_t)BB * LL * DD;        // 8.4M
  __bf16* wqt  = p; p += (size_t)HH * DD * DKK;       // 1.05M
  __bf16* wkt  = p; p += (size_t)HH * DD * DKK;
  __bf16* wvt  = p; p += (size_t)HH * DD * DKK;
  __bf16* wot  = p; p += (size_t)DD * HDIM;           // 1.05M
  __bf16* qg   = p; p += (size_t)BB * HH * LL * DKK;  // 8.4M
  __bf16* kg   = p; p += (size_t)BB * HH * LL * DKK;
  __bf16* vtg  = p; p += (size_t)BB * HH * LL * DKK;
  __bf16* hds  = p; p += (size_t)BB * LL * HDIM;      // 8.4M

  k_transpose_x<<<(BB * DD * LL) / 256, 256, 0, stream>>>(x, xt);
  k_convert_wt<<<(HH * DD * DKK) / 256, 256, 0, stream>>>(Wq, wqt, DD, DKK);
  k_convert_wt<<<(HH * DD * DKK) / 256, 256, 0, stream>>>(Wk, wkt, DD, DKK);
  k_convert_wt<<<(HH * DD * DKK) / 256, 256, 0, stream>>>(Wv, wvt, DD, DKK);
  k_convert_wt<<<(DD * HDIM) / 256, 256, 0, stream>>>(Wo, wot, DD, HDIM);

  k_qkv<<<dim3(LL / 64, BB * HH), 256, 0, stream>>>(xt, wqt, wkt, wvt, qg, kg, vtg);
  k_attn<<<dim3(LL / 64, BB * HH), 128, 0, stream>>>(qg, kg, vtg, mask, hds);
  k_out<<<dim3(LL / 64, HDIM / 128, BB), 256, 0, stream>>>(hds, wot, out);
}
